// Mamba2_72473278152756
// MI455X (gfx1250) — matrix operand order repaired and checked by reading, not compiled, not run on hardware
//
#include <hip/hip_runtime.h>
#include <hip/hip_bf16.h>
#include <cstdint>
#include <cstddef>

// ---------------------------------------------------------------------------
// Mamba2 layer for MI455X (gfx1250):
//   bf16 WMMA GEMMs fed by GLOBAL_LOAD_ASYNC_TO_LDS_B128 (ASYNCcnt pipeline)
//   + 3-pass chunked selective scan (sequential depth 4096 -> 256).
// ---------------------------------------------------------------------------

typedef __attribute__((ext_vector_type(16))) __bf16 v16bf;
typedef __attribute__((ext_vector_type(8)))  float  v8f;

#define BB      2
#define LL      4096
#define DMODEL  2048
#define DINNER  4096
#define NH      64
#define HD      64
#define MTOT    (BB * LL)          // 8192 rows
#define SEG     16                 // scan segments
#define SEGLEN  (LL / SEG)         // 256 steps per segment

// ---- bf16 helpers (round-to-nearest-even down-convert) --------------------
__device__ __forceinline__ unsigned short f2b(float f) {
  unsigned u = __float_as_uint(f);
  unsigned r = u + 0x7FFFu + ((u >> 16) & 1u);
  return (unsigned short)(r >> 16);
}
__device__ __forceinline__ float b2f(unsigned short h) {
  return __uint_as_float(((unsigned)h) << 16);
}

// ---------------------------------------------------------------------------
// K0: f32 -> bf16 cast (grid-stride)
// ---------------------------------------------------------------------------
__global__ void cast_f32_to_bf16(const float* __restrict__ src,
                                 unsigned short* __restrict__ dst, int n) {
  int i = blockIdx.x * blockDim.x + threadIdx.x;
  int stride = gridDim.x * blockDim.x;
  for (; i < n; i += stride) dst[i] = f2b(src[i]);
}

// ---------------------------------------------------------------------------
// K1: tiled bf16 WMMA GEMM:  C[M,N] = A[M,K] * W[N,K]^T
//   MODE 0: store bf16 (proj)   MODE 1: sigmoid->bf16 (gate)   MODE 2: f32
// Block 256 threads (8 waves). Tile 128x128, K-slab 64 (2 WMMA K-steps).
// Double-buffered LDS stages filled by global_load_async_to_lds_b128;
// synchronized with s_wait_asynccnt + workgroup barrier.
// ---------------------------------------------------------------------------
template <int MODE>
__launch_bounds__(256)
__global__ void gemm_bf16_wmma(const unsigned short* __restrict__ A,
                               const unsigned short* __restrict__ W,
                               void* __restrict__ Cout,
                               int M, int N, int K) {
  constexpr int BK     = 64;
  constexpr int LSTR   = 72;            // 64 + 8 pad elements (16B-aligned rows)
  constexpr int TILE_E = 128 * LSTR;    // 9216 u16 per tile
  constexpr int STAGE_E = 2 * TILE_E;   // A tile + B tile per stage
  __shared__ __align__(16) unsigned short smem[2 * STAGE_E];  // 73,728 B

  const int tid = threadIdx.x;
  const int m0  = blockIdx.y * 128;
  const int n0  = blockIdx.x * 128;

  // wave compute mapping: 4 m-waves x 2 n-waves, each wave -> 32x64 of C
  const int wid  = tid >> 5;
  const int lane = tid & 31;
  const int lm   = lane & 15;
  const int lh   = lane >> 4;           // half-wave select
  const int wm   = (wid & 3) * 32;
  const int wn   = (wid >> 2) * 64;

  // async staging: 128 rows x 64 cols bf16 = 1024 16B chunks per tile;
  // each thread issues 4 chunks for A and 4 for B per slab (8 asyncs total).
  auto issue_tile = [&](const unsigned short* src, int rowBase, int kt, int stage,
                        int tileOff) {
    for (int i = 0; i < 4; ++i) {
      int chunk = tid + 256 * i;                 // 0..1023
      int row = chunk >> 3;                      // 0..127
      int cx  = chunk & 7;                       // 16B chunk within row
      unsigned ldsOff =
          (unsigned)((stage * STAGE_E + tileOff + row * LSTR + cx * 8) * 2);
      unsigned long long ga = (unsigned long long)(uintptr_t)(
          src + (size_t)(rowBase + row) * K + kt * BK + cx * 8);
      asm volatile("global_load_async_to_lds_b128 %0, %1, off"
                   :: "v"(ldsOff), "v"(ga) : "memory");
    }
  };

  v8f acc[2][4];
  for (int i = 0; i < 2; ++i)
    for (int j = 0; j < 4; ++j)
      for (int r = 0; r < 8; ++r) acc[i][j][r] = 0.0f;

  const int KT = K / BK;

  // prologue: slab 0 into stage 0
  issue_tile(A, m0, 0, 0, 0);
  issue_tile(W, n0, 0, 0, TILE_E);

  for (int kt = 0; kt < KT; ++kt) {
    const int cur = kt & 1;
    if (kt + 1 < KT) {
      issue_tile(A, m0, kt + 1, cur ^ 1, 0);
      issue_tile(W, n0, kt + 1, cur ^ 1, TILE_E);
      // 8 asyncs outstanding for slab kt+1; wait for slab kt's 8 to land
      asm volatile("s_wait_asynccnt 0x8" ::: "memory");
    } else {
      asm volatile("s_wait_asynccnt 0x0" ::: "memory");
    }
    __syncthreads();                              // stage `cur` visible to all

    const unsigned short* sA = &smem[cur * STAGE_E];
    const unsigned short* sB = &smem[cur * STAGE_E + TILE_E];

    for (int ks = 0; ks < 2; ++ks) {              // two 16x16x32 K-steps
      const int koff = ks * 32;
      union Frag { v16bf v; uint4 q[2]; };
      Frag fa[2], fb[4];
      for (int i = 0; i < 2; ++i) {
        // A 16x32 layout: lane%16 = M; lane<16 -> K {0..7,16..23} else {8..15,24..31}
        const unsigned short* p = &sA[(wm + i * 16 + lm) * LSTR + koff + lh * 8];
        fa[i].q[0] = *(const uint4*)p;
        fa[i].q[1] = *(const uint4*)(p + 16);
      }
      for (int j = 0; j < 4; ++j) {
        // B 32x16 layout: lane%16 = N; lane<16 -> K {0..7,16..23} else {8..15,24..31}
        const unsigned short* p = &sB[(wn + j * 16 + lm) * LSTR + koff + lh * 8];
        fb[j].q[0] = *(const uint4*)p;
        fb[j].q[1] = *(const uint4*)(p + 16);
      }
      for (int i = 0; i < 2; ++i)
        for (int j = 0; j < 4; ++j)
          acc[i][j] = __builtin_amdgcn_wmma_f32_16x16x32_bf16(
              false, fa[i].v, false, fb[j].v, (short)0, acc[i][j], false, false);
    }

    __syncthreads();   // protect stage `cur` before it is re-filled at kt+2
  }

  // epilogue: C 16x16 f32 layout -> VGPR r, lanes0-15: M=r, lanes16-31: M=r+8
  for (int i = 0; i < 2; ++i)
    for (int j = 0; j < 4; ++j)
      for (int r = 0; r < 8; ++r) {
        int row = m0 + wm + i * 16 + r + 8 * lh;
        int col = n0 + wn + j * 16 + lm;
        float v = acc[i][j][r];
        if (MODE == 0) {
          ((unsigned short*)Cout)[(size_t)row * N + col] = f2b(v);
        } else if (MODE == 1) {
          float s = 1.0f / (1.0f + __expf(-v));
          ((unsigned short*)Cout)[(size_t)row * N + col] = f2b(s);
        } else {
          ((float*)Cout)[(size_t)row * N + col] = v;
        }
      }
}

// ---------------------------------------------------------------------------
// K2: depthwise causal conv (width 4) + SiLU.  thread per (b,t,c)
// ---------------------------------------------------------------------------
__global__ void conv_silu_kernel(const unsigned short* __restrict__ proj,
                                 const float* __restrict__ conv_w,
                                 const float* __restrict__ conv_b,
                                 unsigned short* __restrict__ Xout) {
  int idx = blockIdx.x * blockDim.x + threadIdx.x;   // < 2*4096*4096 = 2^25
  int c = idx & 4095;
  int t = (idx >> 12) & 4095;
  int b = idx >> 24;
  float4 w = *(const float4*)&conv_w[c * 4];
  float acc = conv_b[c];
  int rowBase = (b << 12);
  if (t >= 3) acc += w.x * b2f(proj[(size_t)(rowBase + t - 3) * DINNER + c]);
  if (t >= 2) acc += w.y * b2f(proj[(size_t)(rowBase + t - 2) * DINNER + c]);
  if (t >= 1) acc += w.z * b2f(proj[(size_t)(rowBase + t - 1) * DINNER + c]);
  acc += w.w * b2f(proj[(size_t)(rowBase + t) * DINNER + c]);
  acc = acc / (1.0f + __expf(-acc));                 // SiLU
  Xout[(size_t)idx] = f2b(acc);
}

// ---------------------------------------------------------------------------
// K3: dt[b,t,h] = softplus( dot(X[b,t,h,:], dt_w[h,:]) + dt_b[h] )
// ---------------------------------------------------------------------------
__global__ void dt_kernel(const unsigned short* __restrict__ X,
                          const float* __restrict__ dt_w,
                          const float* __restrict__ dt_b,
                          float* __restrict__ dt_out) {
  int id = blockIdx.x * blockDim.x + threadIdx.x;    // < MTOT*NH = 524288
  int h  = id & 63;
  int bt = id >> 6;
  const unsigned short* xp = X + (size_t)bt * DINNER + h * HD;
  const float* wp = dt_w + h * HD;
  float acc = dt_b[h];
  for (int i = 0; i < 8; ++i) {
    union { uint4 q; unsigned short s[8]; } u;
    u.q = *(const uint4*)(xp + i * 8);
    for (int j = 0; j < 8; ++j) acc += b2f(u.s[j]) * wp[i * 8 + j];
  }
  dt_out[id] = (acc > 20.0f) ? acc : log1pf(__expf(acc));
}

// ---------------------------------------------------------------------------
// K4a/b/c: chunked selective scan.
//   s_t = d_t*s_{t-1} + u_t,  d = exp(dt*A),  u = dt*B*x
// pass1: per (b,ch,seg): segment decay product P and zero-start end state E
// pass2: per (b,ch): 16-step carry across segments -> carry-in C per segment
// pass3: per (b,ch,seg): replay from carry-in, fuse y = Cp*s + Dp*x and gate
// index layout: id = ((b*SEG + g) * DINNER) + ch   (ch fastest -> coalesced)
// ---------------------------------------------------------------------------
__global__ void scan_pass1(const unsigned short* __restrict__ X,
                           const float* __restrict__ dt,
                           const float* __restrict__ A_log,
                           const float* __restrict__ Bp,
                           float* __restrict__ Pg, float* __restrict__ Eg) {
  int id = blockIdx.x * blockDim.x + threadIdx.x;    // < BB*SEG*DINNER = 131072
  int ch = id & 4095;
  int g  = (id >> 12) & (SEG - 1);
  int b  = id >> 16;
  int h  = ch >> 6;
  float Av = -__expf(A_log[ch]);
  float Bv = Bp[ch];
  float P = 1.0f, s = 0.0f;
  int rowBase = (b << 12) + g * SEGLEN;
  for (int t = 0; t < SEGLEN; ++t) {
    int row = rowBase + t;
    float x   = b2f(X[(size_t)row * DINNER + ch]);
    float dtv = dt[row * NH + h];
    float d   = __expf(dtv * Av);
    s = d * s + dtv * Bv * x;
    P *= d;
  }
  Pg[id] = P;
  Eg[id] = s;
}

__global__ void scan_pass2(const float* __restrict__ Pg,
                           const float* __restrict__ Eg,
                           float* __restrict__ Cg) {
  int id = blockIdx.x * blockDim.x + threadIdx.x;    // < BB*DINNER = 8192
  int ch = id & 4095;
  int b  = id >> 12;
  float s = 0.0f;
  for (int g = 0; g < SEG; ++g) {
    int idx = (((b * SEG + g) << 12)) + ch;
    Cg[idx] = s;                   // state entering segment g
    s = Eg[idx] + Pg[idx] * s;
  }
}

__global__ void scan_pass3(const unsigned short* __restrict__ X,
                           const unsigned short* __restrict__ gate,
                           const float* __restrict__ dt,
                           const float* __restrict__ A_log,
                           const float* __restrict__ Bp,
                           const float* __restrict__ Cp,
                           const float* __restrict__ Dp,
                           const float* __restrict__ Cg,
                           unsigned short* __restrict__ comb) {
  int id = blockIdx.x * blockDim.x + threadIdx.x;    // < 131072
  int ch = id & 4095;
  int g  = (id >> 12) & (SEG - 1);
  int b  = id >> 16;
  int h  = ch >> 6;
  float Av = -__expf(A_log[ch]);
  float Bv = Bp[ch], Cv = Cp[ch], Dv = Dp[ch];
  float s = Cg[id];
  int rowBase = (b << 12) + g * SEGLEN;
  for (int t = 0; t < SEGLEN; ++t) {
    int row = rowBase + t;
    float x   = b2f(X[(size_t)row * DINNER + ch]);
    float dtv = dt[row * NH + h];
    s = __expf(dtv * Av) * s + dtv * Bv * x;
    float y = Cv * s + Dv * x;
    float gv = b2f(gate[(size_t)row * DINNER + ch]);
    comb[(size_t)row * DINNER + ch] = f2b(y * gv);
  }
}

// ---------------------------------------------------------------------------
// Host-side launch sequence (all on `stream`; workspace carved from d_ws).
// ---------------------------------------------------------------------------
extern "C" void kernel_launch(void* const* d_in, const int* in_sizes, int n_in,
                              void* d_out, int out_size, void* d_ws, size_t ws_size,
                              hipStream_t stream) {
  const float* hidden = (const float*)d_in[0];
  const float* Wi     = (const float*)d_in[1];
  const float* Wg     = (const float*)d_in[2];
  const float* Wo     = (const float*)d_in[3];
  const float* conv_w = (const float*)d_in[4];
  const float* conv_b = (const float*)d_in[5];
  const float* dt_w   = (const float*)d_in[6];
  const float* dt_b   = (const float*)d_in[7];
  const float* A_log  = (const float*)d_in[8];
  const float* Bpp    = (const float*)d_in[9];
  const float* Cpp    = (const float*)d_in[10];
  const float* Dpp    = (const float*)d_in[11];
  float* out = (float*)d_out;

  // ---- workspace layout ---------------------------------------------------
  char* ws = (char*)d_ws;
  size_t off = 0;
  unsigned short* X_bf  = (unsigned short*)(ws + off); off += (size_t)MTOT * DMODEL * 2;
  unsigned short* Wi_bf = (unsigned short*)(ws + off); off += (size_t)DINNER * DMODEL * 2;
  unsigned short* Wg_bf = (unsigned short*)(ws + off); off += (size_t)DINNER * DMODEL * 2;
  unsigned short* Wo_bf = (unsigned short*)(ws + off); off += (size_t)DMODEL * DINNER * 2;
  unsigned short* proj  = (unsigned short*)(ws + off); off += (size_t)MTOT * DINNER * 2;
  unsigned short* gateb = (unsigned short*)(ws + off); off += (size_t)MTOT * DINNER * 2;
  unsigned short* xconv = (unsigned short*)(ws + off); off += (size_t)MTOT * DINNER * 2;
  float*          dtbuf = (float*)(ws + off);          off += (size_t)MTOT * NH * 4;
  float*          Pg    = (float*)(ws + off);          off += (size_t)BB * SEG * DINNER * 4;
  float*          Eg    = (float*)(ws + off);          off += (size_t)BB * SEG * DINNER * 4;
  float*          Cg    = (float*)(ws + off);          off += (size_t)BB * SEG * DINNER * 4;
  unsigned short* comb  = proj;  // proj dead after conv: reuse for combined

  // ---- 1) f32 -> bf16 casts ----------------------------------------------
  cast_f32_to_bf16<<<4096, 256, 0, stream>>>(hidden, X_bf,  MTOT * DMODEL);
  cast_f32_to_bf16<<<4096, 256, 0, stream>>>(Wi,     Wi_bf, DINNER * DMODEL);
  cast_f32_to_bf16<<<4096, 256, 0, stream>>>(Wg,     Wg_bf, DINNER * DMODEL);
  cast_f32_to_bf16<<<4096, 256, 0, stream>>>(Wo,     Wo_bf, DMODEL * DINNER);

  // ---- 2) proj / gate GEMMs (bf16 WMMA, async-to-LDS pipeline) ------------
  dim3 gProj(DINNER / 128, MTOT / 128);   // (32, 64)
  gemm_bf16_wmma<0><<<gProj, 256, 0, stream>>>(X_bf, Wi_bf, proj,  MTOT, DINNER, DMODEL);
  gemm_bf16_wmma<1><<<gProj, 256, 0, stream>>>(X_bf, Wg_bf, gateb, MTOT, DINNER, DMODEL);

  // ---- 3) depthwise conv + SiLU ------------------------------------------
  conv_silu_kernel<<<(MTOT * DINNER) / 256, 256, 0, stream>>>(proj, conv_w, conv_b, xconv);

  // ---- 4) dt projection + softplus ---------------------------------------
  dt_kernel<<<(MTOT * NH) / 256, 256, 0, stream>>>(xconv, dt_w, dt_b, dtbuf);

  // ---- 5) chunked selective scan + gating -> combined ---------------------
  scan_pass1<<<(BB * SEG * DINNER) / 256, 256, 0, stream>>>(xconv, dtbuf, A_log, Bpp, Pg, Eg);
  scan_pass2<<<(BB * DINNER) / 256, 256, 0, stream>>>(Pg, Eg, Cg);
  scan_pass3<<<(BB * SEG * DINNER) / 256, 256, 0, stream>>>(xconv, gateb, dtbuf,
                                                            A_log, Bpp, Cpp, Dpp, Cg, comb);

  // ---- 6) out projection (bf16 WMMA, f32 store) ---------------------------
  dim3 gOut(DMODEL / 128, MTOT / 128);    // (16, 64)
  gemm_bf16_wmma<2><<<gOut, 256, 0, stream>>>(comb, Wo_bf, out, MTOT, DMODEL, DINNER);

  (void)in_sizes; (void)n_in; (void)out_size; (void)ws_size;
}
